// FrameTransformerEncoder_6622839570819
// MI455X (gfx1250) — matrix operand order repaired and checked by reading, not compiled, not run on hardware
//
#include <hip/hip_runtime.h>
#include <hip/hip_bf16.h>
#include <math.h>

// ---------------- problem dims (compile-time) ----------------
#define BB   4
#define NCIN 2
#define CC   8
#define FF   1024
#define WFR  512          // frames (the GEMM N dimension)
#define HH   8
#define DH   128          // F / H
#define EE   4096
#define EPSLN 1e-5f

typedef __attribute__((ext_vector_type(16))) __bf16 v16bf;
typedef __attribute__((ext_vector_type(8)))  float  v8f;

// fp32 -> bf16 (hardware cvt; gfx1250 packs these as v_cvt_pk_bf16_f32)
__device__ __forceinline__ __bf16 f2bf(float f) { return (__bf16)f; }

__device__ __forceinline__ unsigned pack2bf(float lo, float hi) {
    unsigned short a = __builtin_bit_cast(unsigned short, f2bf(lo));
    unsigned short b = __builtin_bit_cast(unsigned short, f2bf(hi));
    return (unsigned)a | ((unsigned)b << 16);
}

// =====================================================================
// Per-channel GEMM:  Y[bc][g][w] (op)= act( sum_f A[c][g][f] * X[bc][f][w] )
//   X: (B*C, K, WFR)   A: (C, G, K)   Y: (B*C, G, WFR)
// block = 256 thr = 8 waves; block tile 128(G) x 64(W); wave tile 16 x 64.
// B chunk (32K x 64N) staged in LDS as bf16, shared by all 8 waves.
// grid = (WFR/64, G/128, B*C)
// =====================================================================
template <int ACT, int ADDTO>
__global__ __launch_bounds__(256) void mc_gemm_wmma(
    const float* __restrict__ X, const float* __restrict__ A,
    float* __restrict__ Y, int G, int K, int C)
{
    __shared__ __bf16 Bs[32][80];          // row stride 160 B (32-B aligned rows)

    const int lane = threadIdx.x & 31;
    const int wave = threadIdx.x >> 5;
    const int bc   = blockIdx.z;
    const int c    = bc % C;
    const int g0   = blockIdx.y * 128 + wave * 16;
    const int w0   = blockIdx.x * 64;

    const float* Xp = X + (size_t)bc * K * WFR;
    const float* Ap = A + (size_t)c * G * K + (size_t)(g0 + (lane & 15)) * K;
    const int khalf = (lane >> 4) * 8;     // A K-run selector per ISA layout

    v8f acc[4] = {v8f{}, v8f{}, v8f{}, v8f{}};

    for (int k0 = 0; k0 < K; k0 += 32) {
        __syncthreads();
        // cooperative load: 32x64 floats -> bf16 LDS, float2 per slot
        #pragma unroll
        for (int i = 0; i < 4; ++i) {
            int e   = (threadIdx.x + 256 * i) * 2;        // even element index
            int kr  = e >> 6, col = e & 63;
            const float2 v = *(const float2*)(Xp + (size_t)(k0 + kr) * WFR + w0 + col);
            *(unsigned*)&Bs[kr][col] = pack2bf(v.x, v.y);
        }
        __syncthreads();

        // A fragment: two 8-wide K runs selected by lane[4]
        v16bf av;
        const float* ap = Ap + k0 + khalf;
        #pragma unroll
        for (int i = 0; i < 8; ++i) {
            av[i]     = f2bf(ap[i]);        // K = k0+khalf+i
            av[i + 8] = f2bf(ap[i + 16]);   // K = k0+khalf+16+i
        }
        if (k0 + 32 < K) __builtin_prefetch(ap + 32, 0, 0);

        #pragma unroll
        for (int j = 0; j < 4; ++j) {
            v16bf bv;
            #pragma unroll
            for (int e = 0; e < 16; ++e) bv[e] = Bs[khalf + (e & 7) + ((e >> 3) << 4)][16 * j + (lane & 15)];
            acc[j] = __builtin_amdgcn_wmma_f32_16x16x32_bf16(
                         false, av, false, bv, (short)0, acc[j], false, false);
        }
    }

    float* Yp = Y + (size_t)bc * G * WFR;
    const int mb = g0 + (lane >> 4) * 8;
    #pragma unroll
    for (int j = 0; j < 4; ++j) {
        const int n = w0 + 16 * j + (lane & 15);
        #pragma unroll
        for (int r = 0; r < 8; ++r) {
            float v = acc[j][r];
            if (ACT) { v = fmaxf(v, 0.f); v *= v; }      // SquaredReLU epilogue
            float* yp = Yp + (size_t)(mb + r) * WFR + n;
            if (ADDTO) *yp += v; else *yp = v;           // fused residual add
        }
    }
}

// =====================================================================
// Attention scores:  qk[bch][w][v] = scale * sum_d Q[d][w] * K[d][v]
// Q,K stored as (B*C, F, WFR); head h uses rows [h*DH, h*DH+DH)
// wave tile 16(w) x 64(v), 4 accumulators; grid = (WFR/64, WFR/128, B*C*H)
// =====================================================================
__global__ __launch_bounds__(256) void attn_scores_wmma(
    const float* __restrict__ Q, const float* __restrict__ Kb,
    float* __restrict__ QKout, float scale)
{
    const int lane = threadIdx.x & 31;
    const int wave = threadIdx.x >> 5;
    const int bch = blockIdx.z;
    const int bc  = bch / HH, hh = bch % HH;
    const int w0  = blockIdx.y * 128 + wave * 16;        // M = query frame
    const int v0  = blockIdx.x * 64;                     // N base = key frame

    const float* Qp = Q  + (size_t)bc * FF * WFR + (size_t)hh * DH * WFR;
    const float* Kp = Kb + (size_t)bc * FF * WFR + (size_t)hh * DH * WFR;
    const int m     = w0 + (lane & 15);
    const int khalf = (lane >> 4) * 8;

    v8f acc[4] = {v8f{}, v8f{}, v8f{}, v8f{}};
    for (int k0 = 0; k0 < DH; k0 += 32) {
        v16bf av;
        #pragma unroll
        for (int i = 0; i < 8; ++i) {                    // A = Q^T (strided)
            av[i]     = f2bf(Qp[(size_t)(k0 + khalf +      i) * WFR + m]);
            av[i + 8] = f2bf(Qp[(size_t)(k0 + khalf + 16 + i) * WFR + m]);
        }
        const float* bp = Kp + (size_t)k0 * WFR + v0 + (lane & 15);   // lane%16 = N col
        #pragma unroll
        for (int j = 0; j < 4; ++j) {
            v16bf bv;
            #pragma unroll
            for (int e = 0; e < 16; ++e) bv[e] = f2bf(bp[(size_t)(khalf + (e & 7) + ((e >> 3) << 4)) * WFR + 16 * j]);
            acc[j] = __builtin_amdgcn_wmma_f32_16x16x32_bf16(
                         false, av, false, bv, (short)0, acc[j], false, false);
        }
    }

    float* out = QKout + (size_t)bch * WFR * WFR;
    const int mb = w0 + (lane >> 4) * 8;
    #pragma unroll
    for (int j = 0; j < 4; ++j) {
        const int n = v0 + 16 * j + (lane & 15);
        #pragma unroll
        for (int r = 0; r < 8; ++r)
            out[(size_t)(mb + r) * WFR + n] = acc[j][r] * scale;
    }
}

// =====================================================================
// Attention output:  O[h*DH+d][w] = sum_v P[w][v] * V[h*DH+d][v]
// P: (B*C*H, WFR, WFR) probs;  V,O: (B*C, F, WFR)
// wave tile 16(w) x 64(d); grid = (DH/64, WFR/128, B*C*H)
// =====================================================================
__global__ __launch_bounds__(256) void attn_out_wmma(
    const float* __restrict__ P, const float* __restrict__ V,
    float* __restrict__ O)
{
    const int lane = threadIdx.x & 31;
    const int wave = threadIdx.x >> 5;
    const int bch = blockIdx.z;
    const int bc  = bch / HH, hh = bch % HH;
    const int w0  = blockIdx.y * 128 + wave * 16;        // M = frame
    const int d0  = blockIdx.x * 64;                     // N base = head dim

    const float* Pp = P + (size_t)bch * WFR * WFR + (size_t)(w0 + (lane & 15)) * WFR;
    const float* Vp = V + (size_t)bc * FF * WFR + (size_t)hh * DH * WFR;
    const int khalf = (lane >> 4) * 8;

    v8f acc[4] = {v8f{}, v8f{}, v8f{}, v8f{}};
    for (int k0 = 0; k0 < WFR; k0 += 32) {
        v16bf av;
        #pragma unroll
        for (int i = 0; i < 8; ++i) {                    // A = P row (contig)
            av[i]     = f2bf(Pp[k0 + khalf +      i]);
            av[i + 8] = f2bf(Pp[k0 + khalf + 16 + i]);
        }
        #pragma unroll
        for (int j = 0; j < 4; ++j) {
            v16bf bv;
            #pragma unroll
            for (int e = 0; e < 16; ++e)                 // B = V^T (strided)
                bv[e] = f2bf(Vp[(size_t)(d0 + 16 * j + (lane & 15)) * WFR + k0 + khalf + (e & 7) + ((e >> 3) << 4)]);
            acc[j] = __builtin_amdgcn_wmma_f32_16x16x32_bf16(
                         false, av, false, bv, (short)0, acc[j], false, false);
        }
    }

    float* Op = O + (size_t)bc * FF * WFR + (size_t)hh * DH * WFR;
    const int mb = w0 + (lane >> 4) * 8;
    #pragma unroll
    for (int j = 0; j < 4; ++j) {
        const int n = d0 + 16 * j + (lane & 15);
        #pragma unroll
        for (int r = 0; r < 8; ++r)
            Op[(size_t)n * WFR + (mb + r)] = acc[j][r];
    }
}

// =====================================================================
// Elementwise / reduction helper kernels
// =====================================================================
__global__ __launch_bounds__(256) void embed_kernel(
    const float* __restrict__ x, const float* __restrict__ ew,
    const float* __restrict__ eb, float* __restrict__ h)
{
    size_t i  = (size_t)blockIdx.x * 256 + threadIdx.x;   // B*C*F*W
    size_t fw = i % ((size_t)FF * WFR);
    int    o  = (int)((i / ((size_t)FF * WFR)) % CC);
    int    b  = (int)( i / ((size_t)FF * WFR * CC));
    float acc = eb[o];
    #pragma unroll
    for (int c = 0; c < NCIN; ++c)
        acc += x[((size_t)(b * NCIN + c)) * FF * WFR + fw] * ew[o * NCIN + c];
    h[i] = acc;
}

// layernorm over F axis, one thread per (bc, w)
__global__ __launch_bounds__(256) void mc_layernorm_kernel(
    const float* __restrict__ X, const float* __restrict__ Wt,
    const float* __restrict__ Bs, float* __restrict__ Y)
{
    int tid = blockIdx.x * 256 + threadIdx.x;             // B*C*WFR threads
    int bc = tid / WFR, w = tid % WFR;
    int c  = bc % CC;
    const float* xp = X + (size_t)bc * FF * WFR + w;
    float s = 0.f, ss = 0.f;
    for (int f = 0; f < FF; ++f) {
        float v = xp[(size_t)f * WFR];
        s += v; ss += v * v;
    }
    float mean = s * (1.0f / FF);
    float var  = ss * (1.0f / FF) - mean * mean;
    float rstd = rsqrtf(var + EPSLN);
    float* yp = Y + (size_t)bc * FF * WFR + w;
    const float* wr = Wt + (size_t)c * FF;
    const float* br = Bs + (size_t)c * FF;
    for (int f = 0; f < FF; ++f)
        yp[(size_t)f * WFR] = (xp[(size_t)f * WFR] - mean) * rstd * wr[f] + br[f];
}

__global__ __launch_bounds__(256) void glu_combine_kernel(
    const float* __restrict__ t, float* __restrict__ h)
{
    size_t i  = (size_t)blockIdx.x * 256 + threadIdx.x;   // B*C*F*W
    size_t fw = i % ((size_t)FF * WFR);
    size_t bc = i / ((size_t)FF * WFR);
    const float* tp = t + bc * 2 * FF * WFR;
    float a = tp[fw];
    float g = tp[(size_t)FF * WFR + fw];
    h[i] += a * (1.0f / (1.0f + __expf(-g)));
}

// depthwise conv along features; Wt: (C,1,KK,1)
template <int KK, int ACT>
__global__ __launch_bounds__(256) void dwconv_feat_kernel(
    const float* __restrict__ X, const float* __restrict__ Wt,
    const float* __restrict__ Bs, float* __restrict__ Y)
{
    size_t i = (size_t)blockIdx.x * 256 + threadIdx.x;    // B*C*F*W
    int w = (int)(i % WFR);
    int f = (int)((i / WFR) % FF);
    int c = (int)((i / ((size_t)WFR * FF)) % CC);
    size_t base = (i / ((size_t)WFR * FF)) * (size_t)WFR * FF;
    const int pad = (KK - 1) / 2;
    float acc = Bs[c];
    #pragma unroll
    for (int t = 0; t < KK; ++t) {
        int fi = f + t - pad;
        if (fi >= 0 && fi < FF)
            acc += X[base + (size_t)fi * WFR + w] * Wt[c * KK + t];
    }
    if (ACT) { acc = fmaxf(acc, 0.f); acc *= acc; }
    Y[i] += acc;                                          // always residual-add
}

// dense conv along frames, K=3; Wt: (C,C,1,3)
__global__ __launch_bounds__(256) void conv_frames_kernel(
    const float* __restrict__ X, const float* __restrict__ Wt,
    const float* __restrict__ Bs, float* __restrict__ Y)
{
    size_t i = (size_t)blockIdx.x * 256 + threadIdx.x;    // B*C*F*W
    int w = (int)(i % WFR);
    int f = (int)((i / WFR) % FF);
    int o = (int)((i / ((size_t)WFR * FF)) % CC);
    int b = (int)( i / ((size_t)WFR * FF * CC));
    const float* xb = X + (size_t)b * CC * FF * WFR;
    float acc = Bs[o];
    #pragma unroll
    for (int c = 0; c < CC; ++c) {
        const float* xr = xb + (size_t)c * FF * WFR + (size_t)f * WFR;
        const float* wr = Wt + ((size_t)o * CC + c) * 3;
        #pragma unroll
        for (int t = 0; t < 3; ++t) {
            int wi = w + t - 1;
            if (wi >= 0 && wi < WFR) acc += xr[wi] * wr[t];
        }
    }
    Y[i] = acc;
}

// row softmax over last axis (WFR), one wave per row
__global__ __launch_bounds__(256) void softmax_rows_kernel(
    const float* __restrict__ X, float* __restrict__ Y, int nrows)
{
    int row  = blockIdx.x * 8 + (threadIdx.x >> 5);
    int lane = threadIdx.x & 31;
    if (row >= nrows) return;
    const float* xp = X + (size_t)row * WFR;
    float mx = -INFINITY;
    for (int i = lane; i < WFR; i += 32) mx = fmaxf(mx, xp[i]);
    #pragma unroll
    for (int o = 16; o > 0; o >>= 1) mx = fmaxf(mx, __shfl_xor(mx, o, 32));
    float s = 0.f;
    for (int i = lane; i < WFR; i += 32) s += __expf(xp[i] - mx);
    #pragma unroll
    for (int o = 16; o > 0; o >>= 1) s += __shfl_xor(s, o, 32);
    float inv = 1.0f / s;
    float* yp = Y + (size_t)row * WFR;
    for (int i = lane; i < WFR; i += 32) yp[i] = __expf(xp[i] - mx) * inv;
}

// out1 = concat(x, h) along channel axis
__global__ __launch_bounds__(256) void concat_out_kernel(
    const float* __restrict__ x, const float* __restrict__ h,
    float* __restrict__ out)
{
    size_t i  = (size_t)blockIdx.x * 256 + threadIdx.x;   // B*(CIN+C)*F*W
    size_t fw = i % ((size_t)FF * WFR);
    int cc = (int)((i / ((size_t)FF * WFR)) % (NCIN + CC));
    int b  = (int)( i / ((size_t)FF * WFR * (NCIN + CC)));
    out[i] = (cc < NCIN)
        ? x[((size_t)(b * NCIN + cc)) * FF * WFR + fw]
        : h[((size_t)(b * CC + (cc - NCIN))) * FF * WFR + fw];
}

__global__ __launch_bounds__(256) void copy_kernel(
    const float* __restrict__ src, float* __restrict__ dst)
{
    size_t i = (size_t)blockIdx.x * 256 + threadIdx.x;
    dst[i] = src[i];
}

// =====================================================================
extern "C" void kernel_launch(void* const* d_in, const int* in_sizes, int n_in,
                              void* d_out, int out_size, void* d_ws, size_t ws_size,
                              hipStream_t stream)
{
    const float* x        = (const float*)d_in[0];
    const float* embed_w  = (const float*)d_in[1];
    const float* embed_b  = (const float*)d_in[2];
    const float* ln1_w    = (const float*)d_in[3];
    const float* ln1_b    = (const float*)d_in[4];
    const float* glu_w    = (const float*)d_in[5];
    const float* ln2_w    = (const float*)d_in[6];
    const float* ln2_b    = (const float*)d_in[7];
    const float* conv1a_w = (const float*)d_in[8];
    const float* conv1b_w = (const float*)d_in[9];
    const float* conv1b_b = (const float*)d_in[10];
    const float* ln3_w    = (const float*)d_in[11];
    const float* ln3_b    = (const float*)d_in[12];
    const float* conv2_w  = (const float*)d_in[13];
    const float* conv2_b  = (const float*)d_in[14];
    const float* ln4_w    = (const float*)d_in[15];
    const float* ln4_b    = (const float*)d_in[16];
    const float* q_lin    = (const float*)d_in[17];
    const float* q_conv_w = (const float*)d_in[18];
    const float* q_conv_b = (const float*)d_in[19];
    const float* k_lin    = (const float*)d_in[20];
    const float* k_conv_w = (const float*)d_in[21];
    const float* k_conv_b = (const float*)d_in[22];
    const float* v_lin    = (const float*)d_in[23];
    const float* v_conv_w = (const float*)d_in[24];
    const float* v_conv_b = (const float*)d_in[25];
    const float* o_lin    = (const float*)d_in[26];
    const float* ln5_w    = (const float*)d_in[27];
    const float* ln5_b    = (const float*)d_in[28];
    const float* conv3_w  = (const float*)d_in[29];
    const float* conv4_w  = (const float*)d_in[30];

    const size_t HW = (size_t)BB * CC * FF * WFR;   // 16,777,216
    float* ws = (float*)d_ws;
    float* buf_h = ws;
    float* buf_z = ws + HW;
    float* buf_q = ws + 2 * HW;
    float* buf_k = ws + 3 * HW;
    float* buf_v = ws + 4 * HW;
    float* buf_o = ws + 5 * HW;
    float* buf_t = ws + 6 * HW;                     // 4*HW (E-sized / 2F / probs)

    float* out1   = (float*)d_out;                          // (B, CIN+C, F, W)
    float* out_h  = out1 + (size_t)BB * (NCIN + CC) * FF * WFR;
    float* out_qk = out_h + HW;                             // (B, C, H, W, W)

    const dim3 blk(256);
    const int  BCi = BB * CC;
    const dim3 g_elt((unsigned)(HW / 256));
    const dim3 g_ln((unsigned)((BB * CC * WFR) / 256));

    // ---- embed: x (B,CIN,F,W) -> h (B,C,F,W)
    embed_kernel<<<g_elt, blk, 0, stream>>>(x, embed_w, embed_b, buf_h);

    // ---- GLU block
    mc_layernorm_kernel<<<g_ln, blk, 0, stream>>>(buf_h, ln1_w, ln1_b, buf_z);
    {   dim3 g(WFR / 64, (2 * FF) / 128, BCi);
        mc_gemm_wmma<0, 0><<<g, blk, 0, stream>>>(buf_z, glu_w, buf_t, 2 * FF, FF, CC); }
    glu_combine_kernel<<<g_elt, blk, 0, stream>>>(buf_t, buf_h);

    // ---- parallel linear + depthwise branch, SquaredReLU, residual
    mc_layernorm_kernel<<<g_ln, blk, 0, stream>>>(buf_h, ln2_w, ln2_b, buf_z);
    {   dim3 g(WFR / 64, FF / 128, BCi);
        mc_gemm_wmma<1, 1><<<g, blk, 0, stream>>>(buf_z, conv1a_w, buf_h, FF, FF, CC); }
    dwconv_feat_kernel<3, 1><<<g_elt, blk, 0, stream>>>(buf_z, conv1b_w, conv1b_b, buf_h);

    // ---- wide depthwise FrameConv (K=9), residual
    mc_layernorm_kernel<<<g_ln, blk, 0, stream>>>(buf_h, ln3_w, ln3_b, buf_z);
    dwconv_feat_kernel<9, 0><<<g_elt, blk, 0, stream>>>(buf_z, conv2_w, conv2_b, buf_h);

    // ---- attention
    mc_layernorm_kernel<<<g_ln, blk, 0, stream>>>(buf_h, ln4_w, ln4_b, buf_z);
    {   dim3 g(WFR / 64, FF / 128, BCi);
        mc_gemm_wmma<0, 0><<<g, blk, 0, stream>>>(buf_z, q_lin, buf_t, FF, FF, CC);
        conv_frames_kernel<<<g_elt, blk, 0, stream>>>(buf_t, q_conv_w, q_conv_b, buf_q);
        mc_gemm_wmma<0, 0><<<g, blk, 0, stream>>>(buf_z, k_lin, buf_t, FF, FF, CC);
        conv_frames_kernel<<<g_elt, blk, 0, stream>>>(buf_t, k_conv_w, k_conv_b, buf_k);
        mc_gemm_wmma<0, 0><<<g, blk, 0, stream>>>(buf_z, v_lin, buf_t, FF, FF, CC);
        conv_frames_kernel<<<g_elt, blk, 0, stream>>>(buf_t, v_conv_w, v_conv_b, buf_v);
    }
    {   dim3 g(WFR / 64, WFR / 128, BCi * HH);
        attn_scores_wmma<<<g, blk, 0, stream>>>(buf_q, buf_k, out_qk,
                                                0.08838834764831845f /* 1/sqrt(128) */); }
    {   int nrows = BB * CC * HH * WFR;
        softmax_rows_kernel<<<dim3(nrows / 8), blk, 0, stream>>>(out_qk, buf_t, nrows); }
    {   dim3 g(DH / 64, WFR / 128, BCi * HH);
        attn_out_wmma<<<g, blk, 0, stream>>>(buf_t, buf_v, buf_o); }
    {   dim3 g(WFR / 64, FF / 128, BCi);
        mc_gemm_wmma<0, 1><<<g, blk, 0, stream>>>(buf_o, o_lin, buf_h, FF, FF, CC); }

    // ---- FFN
    mc_layernorm_kernel<<<g_ln, blk, 0, stream>>>(buf_h, ln5_w, ln5_b, buf_z);
    {   dim3 g(WFR / 64, EE / 128, BCi);
        mc_gemm_wmma<1, 0><<<g, blk, 0, stream>>>(buf_z, conv3_w, buf_t, EE, FF, CC); }
    {   dim3 g(WFR / 64, FF / 128, BCi);
        mc_gemm_wmma<0, 1><<<g, blk, 0, stream>>>(buf_t, conv4_w, buf_h, FF, EE, CC); }

    // ---- outputs: (concat(x,h), h, qk)
    {   size_t n1 = (size_t)BB * (NCIN + CC) * FF * WFR;
        concat_out_kernel<<<dim3((unsigned)(n1 / 256)), blk, 0, stream>>>(x, buf_h, out1); }
    copy_kernel<<<g_elt, blk, 0, stream>>>(buf_h, out_h);
}
